// RNNDecoder_46952582479885
// MI455X (gfx1250) — matrix operand order repaired and checked by reading, not compiled, not run on hardware
//
#include <hip/hip_runtime.h>
#include <hip/hip_bf16.h>

// ---------------------------------------------------------------------------
// RNN graph decoder for MI455X (gfx1250), bf16 WMMA pipeline.
// B=512, N=50, LATENT=128, HID=256, 3H=768, HEADS=4 (hd=64), P=1225.
// ---------------------------------------------------------------------------

#define BV    512
#define NNODE 50
#define LAT   128
#define HIDD  256
#define G3    768
#define PAIRS 1225

typedef __attribute__((ext_vector_type(16))) __bf16 v16bf;
typedef __attribute__((ext_vector_type(8)))  __bf16 v8bf;
typedef __attribute__((ext_vector_type(8)))  float  v8f;

__device__ __forceinline__ __bf16 f2bf(float f) {
    unsigned int u = __builtin_bit_cast(unsigned int, f);
    unsigned int r = (u + 0x7FFFu + ((u >> 16) & 1u)) >> 16;
    unsigned short s = (unsigned short)r;
    return __builtin_bit_cast(__bf16, s);
}
__device__ __forceinline__ float bf2f(__bf16 b) {
    unsigned short s = __builtin_bit_cast(unsigned short, b);
    unsigned int u = ((unsigned int)s) << 16;
    return __builtin_bit_cast(float, u);
}

// A fragment: 16x32 bf16 tile, row-major source with row stride ldk (elems).
// ISA 7.12.2 (16-bit A 16x32): lane<16 -> M=lane, K in {0..7,16..23};
// lane>=16 -> M=lane-16, K in {8..15,24..31}. Two contiguous 16B chunks.
__device__ __forceinline__ v16bf load_frag_A(const __bf16* tile, int ldk) {
    int lane = threadIdx.x & 31;
    const __bf16* p = tile + (lane & 15) * ldk + ((lane >> 4) << 3);
    v8bf lo = *(const v8bf*)(p);
    v8bf hi = *(const v8bf*)(p + 16);
    v16bf f;
#pragma unroll
    for (int e = 0; e < 8; ++e) { f[e] = lo[e]; f[e + 8] = hi[e]; }
    return f;
}

// B fragment: 32x16 bf16 (KxN) where B[k][n] = W[n0+n][k0+k], W row-major
// with row stride ldk. lane<16 -> N=lane, K in {0..7,16..23}; lane>=16 -> N=lane-16,
// K in {8..15,24..31}; two contiguous 16B chunks of a W row (matches A).
__device__ __forceinline__ v16bf load_frag_B(const __bf16* wtile, int ldk) {
    int lane = threadIdx.x & 31;
    const __bf16* p = wtile + (lane & 15) * ldk + ((lane >> 4) << 3);
    v8bf lo = *(const v8bf*)(p);
    v8bf hi = *(const v8bf*)(p + 16);
    v16bf f;
#pragma unroll
    for (int e = 0; e < 8; ++e) { f[e] = lo[e]; f[e + 8] = hi[e]; }
    return f;
}

#define WMMA_BF16(a, b, c) \
    __builtin_amdgcn_wmma_f32_16x16x32_bf16(false, (a), false, (b), (short)0, (c), false, false)

// ---------------------------------------------------------------------------
__global__ void cvt_bf16_kernel(const float* __restrict__ src,
                                __bf16* __restrict__ dst, int n) {
    int i = blockIdx.x * blockDim.x + threadIdx.x;
    if (i < n) dst[i] = f2bf(src[i]);
}

__global__ void zero_f32_kernel(float* __restrict__ p, int n) {
    int i = blockIdx.x * blockDim.x + threadIdx.x;
    if (i < n) p[i] = 0.0f;
}

// seq[b][t][k] = (z[b][k] + pos[t][k]) * (t < n_nodes[b])  -> bf16
__global__ void seq_kernel(const float* __restrict__ z,
                           const float* __restrict__ pos,
                           const int* __restrict__ nn,
                           __bf16* __restrict__ seqB) {
    int idx = blockIdx.x * blockDim.x + threadIdx.x;
    if (idx >= BV * NNODE * LAT) return;
    int k = idx % LAT;
    int t = (idx / LAT) % NNODE;
    int b = idx / (LAT * NNODE);
    float v = (z[b * LAT + k] + pos[t * LAT + k]);
    v = (t < nn[b]) ? v : 0.0f;
    seqB[idx] = f2bf(v);
}

__global__ void pair_table_kernel(int* __restrict__ tab) {
    int p = blockIdx.x * blockDim.x + threadIdx.x;
    if (p >= PAIRS) return;
    int i = 0, rem = p;
    while (rem >= (NNODE - 1 - i)) { rem -= (NNODE - 1 - i); ++i; }
    tab[2 * p] = i;
    tab[2 * p + 1] = i + 1 + rem;
}

// ---------------------------------------------------------------------------
// bf16 GEMM: Y[M][Nout] = A[M][K] @ W[Nout][K]^T + bias.
// Block = 256 thr (8 waves) arranged 2(m) x 4(n); each wave holds a 2x4 grid
// of 16x16 accumulators (32x64 wave tile, 64x256 block tile). Per k-step:
// 2 A-frag loads + 4 B-frag loads feed 8 WMMAs (A reused x4, B reused x2).
// ---------------------------------------------------------------------------
__global__ void gemm_bf16_kernel(const __bf16* __restrict__ A,
                                 const __bf16* __restrict__ W,
                                 const float* __restrict__ bias,
                                 float* __restrict__ Yf,
                                 __bf16* __restrict__ Yb,
                                 int M, int Nout, int K) {
    int wave = threadIdx.x >> 5;
    int lane = threadIdx.x & 31;
    int m0 = blockIdx.x * 64 + (wave >> 2) * 32;   // wave: rows m0..m0+31
    int n0 = blockIdx.y * 256 + (wave & 3) * 64;   // wave: cols n0..n0+63
    if (m0 >= M) return;
    v8f c[2][4] = {};
    for (int kk = 0; kk < K; kk += 32) {
        v16bf a0 = load_frag_A(A + (size_t)m0 * K + kk, K);
        v16bf a1 = load_frag_A(A + (size_t)(m0 + 16) * K + kk, K);
        v16bf b0 = load_frag_B(W + (size_t)(n0 + 0)  * K + kk, K);
        v16bf b1 = load_frag_B(W + (size_t)(n0 + 16) * K + kk, K);
        v16bf b2 = load_frag_B(W + (size_t)(n0 + 32) * K + kk, K);
        v16bf b3 = load_frag_B(W + (size_t)(n0 + 48) * K + kk, K);
        c[0][0] = WMMA_BF16(a0, b0, c[0][0]);
        c[0][1] = WMMA_BF16(a0, b1, c[0][1]);
        c[0][2] = WMMA_BF16(a0, b2, c[0][2]);
        c[0][3] = WMMA_BF16(a0, b3, c[0][3]);
        c[1][0] = WMMA_BF16(a1, b0, c[1][0]);
        c[1][1] = WMMA_BF16(a1, b1, c[1][1]);
        c[1][2] = WMMA_BF16(a1, b2, c[1][2]);
        c[1][3] = WMMA_BF16(a1, b3, c[1][3]);
    }
    int mb = (lane < 16) ? 0 : 8;
#pragma unroll
    for (int i = 0; i < 2; ++i) {
#pragma unroll
        for (int j = 0; j < 4; ++j) {
            int n = n0 + j * 16 + (lane & 15);
            float bv = bias ? bias[n] : 0.0f;
#pragma unroll
            for (int r = 0; r < 8; ++r) {
                float v = c[i][j][r] + bv;
                size_t o = (size_t)(m0 + i * 16 + mb + r) * Nout + n;
                if (Yf) Yf[o] = v;
                if (Yb) Yb[o] = f2bf(v);
            }
        }
    }
}

// ---------------------------------------------------------------------------
// GRU scan over t for one layer. One block owns 16 batch rows, loops t=0..49.
// Per step: the 16x768 f32 xg tile for step t is prefetched into LDS with
// GLOBAL_LOAD_ASYNC_TO_LDS_B128 (ASYNCcnt), overlapping the recurrent WMMA
// GEMM hg = h @ Whh^T + bhh; each wave drains its own ASYNCcnt before the
// barrier preceding the gate math.
// ---------------------------------------------------------------------------
__global__ void gru_scan_kernel(const float* __restrict__ xg,     // [B][50][768]
                                const __bf16* __restrict__ Whh,   // [768][256]
                                const float* __restrict__ bhh,    // [768]
                                __bf16* __restrict__ Hout,        // [B][50][256]
                                const int* __restrict__ nn,
                                int apply_mask) {
    __shared__ __align__(16) __bf16 hS[16 * HIDD];    // 8 KB
    __shared__ __align__(16) float  hgS[16 * G3];     // 48 KB
    __shared__ __align__(16) float  xbuf[16 * G3];    // 48 KB (async dest)
    int tid  = threadIdx.x;
    int wave = tid >> 5;
    int lane = tid & 31;
    int b0   = blockIdx.x * 16;

    // LDS byte offset of xbuf (generic LDS addr: low 32 bits = LDS offset).
    unsigned xbase = (unsigned)(uintptr_t)(void*)xbuf;

    for (int i = tid; i < 16 * HIDD; i += 256) hS[i] = f2bf(0.0f);
    __syncthreads();

    for (int t = 0; t < NNODE; ++t) {
        // ---- async prefetch of xg[:,t,:] tile into LDS (12 b128 per lane)
#pragma unroll
        for (int i = 0; i < 12; ++i) {
            int l = tid + i * 256;           // 0..3071 -> (m, 4-float chunk)
            int m = l / 192;
            int col = (l % 192) * 4;
            unsigned lds_off = xbase + (unsigned)((m * G3 + col) * 4);
            unsigned long long ga =
                (unsigned long long)(const void*)(xg + ((size_t)(b0 + m) * NNODE + t) * G3 + col);
            asm volatile("global_load_async_to_lds_b128 %0, %1, off"
                         :: "v"(lds_off), "v"(ga) : "memory");
        }

        // ---- hg = h @ Whh^T : 48 n-subtiles across 8 waves (6 each), K=256.
#pragma unroll
        for (int s = 0; s < 6; ++s) {
            int n0 = (wave * 6 + s) * 16;
            v8f c = {};
#pragma unroll
            for (int kk = 0; kk < HIDD; kk += 32) {
                v16bf a = load_frag_A(hS + kk, HIDD);
                v16bf b = load_frag_B(Whh + (size_t)n0 * HIDD + kk, HIDD);
                c = WMMA_BF16(a, b, c);
            }
            int n  = n0 + (lane & 15);
            int mb = (lane < 16) ? 0 : 8;
            float bv = bhh[n];
#pragma unroll
            for (int r = 0; r < 8; ++r) hgS[(mb + r) * G3 + n] = c[r] + bv;
        }
        // each wave waits for its own async loads, then block-wide barrier
        asm volatile("s_wait_asynccnt 0x0" ::: "memory");
        __syncthreads();

        // ---- gate update: each thread owns 16 (m, j) elements.
        for (int idx = tid; idx < 16 * HIDD; idx += 256) {
            int m = idx >> 8;
            int j = idx & 255;
            float xr = xbuf[m * G3 + j];
            float xz = xbuf[m * G3 + j + 256];
            float xn = xbuf[m * G3 + j + 512];
            float hr = hgS[m * G3 + j];
            float hz = hgS[m * G3 + j + 256];
            float hn = hgS[m * G3 + j + 512];
            float h  = bf2f(hS[m * HIDD + j]);
            float r  = 1.0f / (1.0f + __expf(-(xr + hr)));
            float zt = 1.0f / (1.0f + __expf(-(xz + hz)));
            float nv = tanhf(xn + r * hn);
            float hnew = (1.0f - zt) * nv + zt * h;
            hS[m * HIDD + j] = f2bf(hnew);
            float o = hnew;
            if (apply_mask && t >= nn[b0 + m]) o = 0.0f;
            Hout[((size_t)(b0 + m) * NNODE + t) * HIDD + j] = f2bf(o);
        }
        __syncthreads();
    }
}

// ---------------------------------------------------------------------------
// Pair kernel: one block = (batch b, 16 pairs). K=2 attention in VALU,
// then two chained 16x256x256 WMMA GEMMs (attn_out, mlp_w1+relu), then the
// binary gumbel straight-through decision and masked symmetric scatter.
// ---------------------------------------------------------------------------
__global__ void pair_kernel(const __bf16* __restrict__ qkv,  // [B][50][768] (q|k|v)
                            const __bf16* __restrict__ Wo,   // [256][256]
                            const float*  __restrict__ bo,   // [256]
                            const __bf16* __restrict__ W1,   // [256][256]
                            const float*  __restrict__ b1,   // [256]
                            const float*  __restrict__ w2,   // [2][256]
                            const float*  __restrict__ b2,   // [2]
                            const float*  __restrict__ gum,  // [B][P][2]
                            const int*    __restrict__ tab,  // [P][2]
                            const int*    __restrict__ nn,   // [B]
                            float* __restrict__ out) {       // [B][50][50]
    __shared__ __align__(16) __bf16 outA[16 * HIDD];  // 8 KB
    __shared__ __align__(16) __bf16 peB[16 * HIDD];   // 8 KB
    __shared__ __align__(16) float  hidS[16 * HIDD];  // 16 KB
    __shared__ float scoreS[16][4][2];
    __shared__ float attnS[16][4][2];
    __shared__ float partial[16][8];
    __shared__ int   piS[16], pjS[16];

    int tid  = threadIdx.x;
    int wave = tid >> 5;
    int lane = tid & 31;
    int b    = blockIdx.x;
    int p0   = blockIdx.y * 16;

    if (tid < 16) {
        int p = p0 + tid;
        if (p >= PAIRS) p = PAIRS - 1;
        piS[tid] = tab[2 * p];
        pjS[tid] = tab[2 * p + 1];
    }
    __syncthreads();

    // scores: 128 threads -> (pair m, head, kv); q from node i, k from i/j.
    if (tid < 128) {
        int m = tid >> 3, s = tid & 7, head = s >> 1, kv = s & 1;
        int ni = piS[m], nk = kv ? pjS[m] : piS[m];
        const __bf16* q = qkv + ((size_t)b * NNODE + ni) * G3 + head * 64;
        const __bf16* k = qkv + ((size_t)b * NNODE + nk) * G3 + 256 + head * 64;
        float acc = 0.0f;
#pragma unroll
        for (int d = 0; d < 64; ++d) acc += bf2f(q[d]) * bf2f(k[d]);
        scoreS[m][head][kv] = acc * 0.125f;  // 1/sqrt(64)
    }
    __syncthreads();
    if (tid < 64) {
        int m = tid >> 2, head = tid & 3;
        float s0 = scoreS[m][head][0], s1 = scoreS[m][head][1];
        float mx = fmaxf(s0, s1);
        float e0 = __expf(s0 - mx), e1 = __expf(s1 - mx);
        float inv = 1.0f / (e0 + e1);
        attnS[m][head][0] = e0 * inv;
        attnS[m][head][1] = e1 * inv;
    }
    __syncthreads();
    {   // out = a_i * v_i + a_j * v_j  (per head), stored bf16 as GEMM A tile
        int m = tid >> 4, c = tid & 15, e0 = c * 16, head = c >> 2;
        float a0 = attnS[m][head][0], a1 = attnS[m][head][1];
        const __bf16* vi = qkv + ((size_t)b * NNODE + piS[m]) * G3 + 512;
        const __bf16* vj = qkv + ((size_t)b * NNODE + pjS[m]) * G3 + 512;
#pragma unroll
        for (int e = e0; e < e0 + 16; ++e)
            outA[m * HIDD + e] = f2bf(a0 * bf2f(vi[e]) + a1 * bf2f(vj[e]));
    }
    __syncthreads();

    // pair_emb = outA @ Wo^T + bo  -> bf16 (16 n-subtiles / 8 waves)
#pragma unroll
    for (int s = 0; s < 2; ++s) {
        int n0 = (wave * 2 + s) * 16;
        v8f c = {};
#pragma unroll
        for (int kk = 0; kk < HIDD; kk += 32) {
            v16bf a = load_frag_A(outA + kk, HIDD);
            v16bf bb = load_frag_B(Wo + (size_t)n0 * HIDD + kk, HIDD);
            c = WMMA_BF16(a, bb, c);
        }
        int n  = n0 + (lane & 15);
        int mb = (lane < 16) ? 0 : 8;
        float bv = bo[n];
#pragma unroll
        for (int r = 0; r < 8; ++r) peB[(mb + r) * HIDD + n] = f2bf(c[r] + bv);
    }
    __syncthreads();

    // hid = relu(pair_emb @ W1^T + b1) -> f32
#pragma unroll
    for (int s = 0; s < 2; ++s) {
        int n0 = (wave * 2 + s) * 16;
        v8f c = {};
#pragma unroll
        for (int kk = 0; kk < HIDD; kk += 32) {
            v16bf a = load_frag_A(peB + kk, HIDD);
            v16bf bb = load_frag_B(W1 + (size_t)n0 * HIDD + kk, HIDD);
            c = WMMA_BF16(a, bb, c);
        }
        int n  = n0 + (lane & 15);
        int mb = (lane < 16) ? 0 : 8;
        float bv = b1[n];
#pragma unroll
        for (int r = 0; r < 8; ++r)
            hidS[(mb + r) * HIDD + n] = fmaxf(c[r] + bv, 0.0f);
    }
    __syncthreads();

    // logit diff (w2 row0 - row1) dot hid
    if (tid < 128) {
        int m = tid >> 3, s = tid & 7, base = s * 32;
        float acc = 0.0f;
#pragma unroll
        for (int d = 0; d < 32; ++d) {
            int e = base + d;
            acc += (w2[e] - w2[256 + e]) * hidS[m * HIDD + e];
        }
        partial[m][s] = acc;
    }
    __syncthreads();
    if (tid < 16) {
        int m = tid, p = p0 + m;
        if (p < PAIRS) {
            float sum = b2[0] - b2[1];
#pragma unroll
            for (int s = 0; s < 8; ++s) sum += partial[m][s];
            const float* g = gum + ((size_t)b * PAIRS + p) * 2;
            float diff = sum + g[0] - g[1];
            float val = (diff >= 0.0f) ? 1.0f : 0.0f;  // hard one-hot, index 0
            int i = piS[m], j = pjS[m];
            int nb = nn[b];
            float v = ((i < nb) && (j < nb)) ? val : 0.0f;
            out[(size_t)b * NNODE * NNODE + i * NNODE + j] = v;
            out[(size_t)b * NNODE * NNODE + j * NNODE + i] = v;
        }
    }
}

// ---------------------------------------------------------------------------
extern "C" void kernel_launch(void* const* d_in, const int* in_sizes, int n_in,
                              void* d_out, int out_size, void* d_ws, size_t ws_size,
                              hipStream_t stream) {
    const float* z       = (const float*)d_in[0];
    const int*   n_nodes = (const int*)d_in[1];
    const float* pos     = (const float*)d_in[3];
    const float* wih0    = (const float*)d_in[4];
    const float* whh0    = (const float*)d_in[5];
    const float* bih0    = (const float*)d_in[6];
    const float* bhh0    = (const float*)d_in[7];
    const float* wih1    = (const float*)d_in[8];
    const float* whh1    = (const float*)d_in[9];
    const float* bih1    = (const float*)d_in[10];
    const float* bhh1    = (const float*)d_in[11];
    const float* nproj   = (const float*)d_in[12];
    const float* attnin  = (const float*)d_in[13];
    const float* attninb = (const float*)d_in[14];
    const float* wo      = (const float*)d_in[15];
    const float* wob     = (const float*)d_in[16];
    const float* w1      = (const float*)d_in[17];
    const float* b1      = (const float*)d_in[18];
    const float* w2      = (const float*)d_in[19];
    const float* b2      = (const float*)d_in[20];
    const float* gum     = (const float*)d_in[21];

    char* ws = (char*)d_ws;
    size_t off = 0;
    auto take = [&](size_t bytes) {
        size_t r = off;
        off += (bytes + 255) & ~(size_t)255;
        return r;
    };
    __bf16* wih0B  = (__bf16*)(ws + take((size_t)G3 * LAT * 2));
    __bf16* whh0B  = (__bf16*)(ws + take((size_t)G3 * HIDD * 2));
    __bf16* wih1B  = (__bf16*)(ws + take((size_t)G3 * HIDD * 2));
    __bf16* whh1B  = (__bf16*)(ws + take((size_t)G3 * HIDD * 2));
    __bf16* nprojB = (__bf16*)(ws + take((size_t)HIDD * HIDD * 2));
    __bf16* atinB  = (__bf16*)(ws + take((size_t)G3 * HIDD * 2));
    __bf16* woB    = (__bf16*)(ws + take((size_t)HIDD * HIDD * 2));
    __bf16* w1B    = (__bf16*)(ws + take((size_t)HIDD * HIDD * 2));
    __bf16* seqB   = (__bf16*)(ws + take((size_t)BV * NNODE * LAT * 2));
    float*  xg     = (float*)(ws + take((size_t)BV * NNODE * G3 * 4));
    __bf16* h0B    = (__bf16*)(ws + take((size_t)BV * NNODE * HIDD * 2));
    __bf16* rnnB   = (__bf16*)(ws + take((size_t)BV * NNODE * HIDD * 2));
    __bf16* nodeB  = (__bf16*)(ws + take((size_t)BV * NNODE * HIDD * 2));
    __bf16* qkvB   = (__bf16*)(ws + take((size_t)BV * NNODE * G3 * 2));
    int*    tab    = (int*)(ws + take((size_t)PAIRS * 2 * 4));
    (void)ws_size; (void)in_sizes; (void)n_in;

    auto cvt = [&](const float* s, __bf16* d, int n) {
        cvt_bf16_kernel<<<(n + 255) / 256, 256, 0, stream>>>(s, d, n);
    };
    cvt(wih0, wih0B, G3 * LAT);
    cvt(whh0, whh0B, G3 * HIDD);
    cvt(wih1, wih1B, G3 * HIDD);
    cvt(whh1, whh1B, G3 * HIDD);
    cvt(nproj, nprojB, HIDD * HIDD);
    cvt(attnin, atinB, G3 * HIDD);
    cvt(wo, woB, HIDD * HIDD);
    cvt(w1, w1B, HIDD * HIDD);

    seq_kernel<<<(BV * NNODE * LAT + 255) / 256, 256, 0, stream>>>(z, pos, n_nodes, seqB);
    pair_table_kernel<<<(PAIRS + 255) / 256, 256, 0, stream>>>(tab);
    zero_f32_kernel<<<(out_size + 255) / 256, 256, 0, stream>>>((float*)d_out, out_size);

    const int M = BV * NNODE;  // 25600
    // layer 0 input projection + scan
    gemm_bf16_kernel<<<dim3(M / 64, G3 / 256), 256, 0, stream>>>(
        seqB, wih0B, bih0, xg, nullptr, M, G3, LAT);
    gru_scan_kernel<<<BV / 16, 256, 0, stream>>>(xg, whh0B, bhh0, h0B, n_nodes, 0);
    // layer 1 input projection + scan (masked output = rnn_out)
    gemm_bf16_kernel<<<dim3(M / 64, G3 / 256), 256, 0, stream>>>(
        h0B, wih1B, bih1, xg, nullptr, M, G3, HIDD);
    gru_scan_kernel<<<BV / 16, 256, 0, stream>>>(xg, whh1B, bhh1, rnnB, n_nodes, 1);
    // per-node projections: node_emb, then fused q|k|v
    gemm_bf16_kernel<<<dim3(M / 64, HIDD / 256), 256, 0, stream>>>(
        rnnB, nprojB, nullptr, nullptr, nodeB, M, HIDD, HIDD);
    gemm_bf16_kernel<<<dim3(M / 64, G3 / 256), 256, 0, stream>>>(
        nodeB, atinB, attninb, nullptr, qkvB, M, G3, HIDD);
    // per-pair attention + MLP + hard decision + symmetric scatter
    pair_kernel<<<dim3(BV, (PAIRS + 15) / 16), 256, 0, stream>>>(
        qkvB, woB, wob, w1B, b1, w2, b2, gum, tab, n_nodes, (float*)d_out);
}
